// SparseMoEFFN_20813411516481
// MI455X (gfx1250) — matrix operand order repaired and checked by reading, not compiled, not run on hardware
//
#include <hip/hip_runtime.h>
#include <hip/hip_bf16.h>
#include <math.h>

// Problem constants (match reference)
#define BB 4
#define TT 2048
#define DD 1024
#define EE 8
#define KK 2
#define HH 2048
#define NTOK (BB*TT)          // 8192 tokens
#define NSLOT (NTOK*KK)       // 16384 (token, slot) rows total

typedef __attribute__((ext_vector_type(16))) __bf16 v16bf;
typedef __attribute__((ext_vector_type(8)))  float  v8f;

union AF { v16bf v; unsigned u[8]; };
union CF { v8f  v; float    f[8]; };

// fp32 -> bf16 bits (RNE) for single values (epilogue only)
__device__ __forceinline__ unsigned short f2bfu(float f) {
    union { __bf16 h; unsigned short s; } c;
    c.h = (__bf16)f;               // native fptrunc, RNE
    return c.s;
}

// pack two fp32 -> bf16x2 (low = a, high = b): exactly one v_cvt_pk_bf16_f32
// (instruction confirmed present on gfx1250 by previous round's disasm)
__device__ __forceinline__ unsigned packbf(float a, float b) {
    unsigned r;
    asm("v_cvt_pk_bf16_f32 %0, %1, %2" : "=v"(r) : "v"(a), "v"(b));
    return r;
}

// ---------------------------------------------------------------------------
// 0) zero per-expert counters (ws is not re-initialized between replays)
// ---------------------------------------------------------------------------
__global__ void moe_init(int* counts, int* cursors) {
    int i = threadIdx.x;
    if (i < EE) { counts[i] = 0; cursors[i] = 0; }
}

// ---------------------------------------------------------------------------
// 1) gating: one wave (32 lanes) per token
// ---------------------------------------------------------------------------
__global__ __launch_bounds__(256) void moe_gate(
    const float* __restrict__ x, const float* __restrict__ Wg,
    const float* __restrict__ bg,
    int* counts, int* topke, float* w0a, float* w1a, float* gate_partial)
{
    const int wid  = threadIdx.x >> 5;
    const int lane = threadIdx.x & 31;
    const int t    = blockIdx.x * 8 + wid;

    const float* xr = x + (size_t)t * DD;
    float acc[EE];
#pragma unroll
    for (int e = 0; e < EE; ++e) acc[e] = 0.f;

    for (int i = lane; i < DD; i += 32) {
        float xv = xr[i];
        const float* wr = Wg + (size_t)i * EE;
#pragma unroll
        for (int e = 0; e < EE; ++e) acc[e] += xv * wr[e];
    }
#pragma unroll
    for (int off = 16; off > 0; off >>= 1) {
#pragma unroll
        for (int e = 0; e < EE; ++e) acc[e] += __shfl_down(acc[e], off, 32);
    }

    __shared__ float pps[8];
    if (lane == 0) {
        float p[EE];
        float m = -1e30f;
#pragma unroll
        for (int e = 0; e < EE; ++e) { p[e] = acc[e] + bg[e]; m = p[e] > m ? p[e] : m; }
        float s = 0.f;
#pragma unroll
        for (int e = 0; e < EE; ++e) { p[e] = expf(p[e] - m); s += p[e]; }
        float inv = 1.f / s;
        float pp = 0.f;
#pragma unroll
        for (int e = 0; e < EE; ++e) { p[e] = 0.99f * p[e] * inv + (0.01f / EE); pp += p[e] * p[e]; }
        // top-2, first occurrence wins on ties (jax top_k convention)
        int i0 = 0; float b0 = p[0];
#pragma unroll
        for (int e = 1; e < EE; ++e) if (p[e] > b0) { b0 = p[e]; i0 = e; }
        int i1 = -1; float b1v = -1.f;
#pragma unroll
        for (int e = 0; e < EE; ++e) if (e != i0 && p[e] > b1v) { b1v = p[e]; i1 = e; }
        float wsum = 1.f / (b0 + b1v);
        topke[t] = i0 | (i1 << 8);
        w0a[t] = b0 * wsum;
        w1a[t] = b1v * wsum;
        atomicAdd(&counts[i0], 1);
        atomicAdd(&counts[i1], 1);
        pps[wid] = pp;
    }
    __syncthreads();
    if (threadIdx.x == 0) {
        float s = 0.f;
#pragma unroll
        for (int i = 0; i < 8; ++i) s += pps[i];
        gate_partial[blockIdx.x] = s;   // deterministic per-block partial
    }
}

// ---------------------------------------------------------------------------
// 2) tiny scan: per-expert offsets + deterministic aux reduction
// ---------------------------------------------------------------------------
__global__ void moe_scan(const int* counts, int* offsets,
                         const float* gate_partial, float* d_aux)
{
    if (threadIdx.x == 0 && blockIdx.x == 0) {
        int s = 0;
        for (int e = 0; e < EE; ++e) { offsets[e] = s; s += counts[e]; }
        float a = 0.f;
        for (int i = 0; i < NTOK / 8; ++i) a += gate_partial[i];
        *d_aux = a * ((float)EE / (float)NTOK);
    }
}

// ---------------------------------------------------------------------------
// 3) build per-expert dispatch lists (packed = token*2 + slot)
// ---------------------------------------------------------------------------
__global__ __launch_bounds__(256) void moe_build(
    const int* topke, const float* w0a, const float* w1a,
    const int* offsets, int* cursors, int* list, float* wlist)
{
    int t = blockIdx.x * blockDim.x + threadIdx.x;
    if (t >= NTOK) return;
    int pk = topke[t];
    int e0 = pk & 0xff, e1 = (pk >> 8) & 0xff;
    int p0 = atomicAdd(&cursors[e0], 1);
    int i0 = offsets[e0] + p0;
    list[i0] = t * 2;     wlist[i0] = w0a[t];
    int p1 = atomicAdd(&cursors[e1], 1);
    int i1 = offsets[e1] + p1;
    list[i1] = t * 2 + 1; wlist[i1] = w1a[t];
}

// ---------------------------------------------------------------------------
// shared helper: convert a 2(K) x 8(N) fp32 block to bf16 K-pairs in Bbuf.
//   r0 = &W[k0 + 2*kp][n0 + seg*8]   (row k even), r1 = r0 + ld (row k odd)
// ---------------------------------------------------------------------------
__device__ __forceinline__ void stage_B_from(const float* r0, const float* r1,
                                             unsigned* dst)
{
    float4 a0 = ((const float4*)r0)[0], a1 = ((const float4*)r0)[1];
    float4 c0 = ((const float4*)r1)[0], c1 = ((const float4*)r1)[1];
    dst[0] = packbf(a0.x, c0.x); dst[1] = packbf(a0.y, c0.y);
    dst[2] = packbf(a0.z, c0.z); dst[3] = packbf(a0.w, c0.w);
    dst[4] = packbf(a1.x, c1.x); dst[5] = packbf(a1.y, c1.y);
    dst[6] = packbf(a1.z, c1.z); dst[7] = packbf(a1.w, c1.w);
}

// ---------------------------------------------------------------------------
// 4) GEMM1: h = gelu(gather(x) @ W1[e] + b1[e])  -> bf16 workspace
//    tile 128(M) x 128(N), K-step 32; wave tile 64x32 (4x2 frags of 16x16)
// ---------------------------------------------------------------------------
__global__ __launch_bounds__(256) void moe_gemm1(
    const float* __restrict__ x, const float* __restrict__ W1,
    const float* __restrict__ b1,
    const int* __restrict__ counts, const int* __restrict__ offsets,
    const int* __restrict__ list, unsigned short* __restrict__ hbuf)
{
    const int e   = blockIdx.z;
    const int cnt = counts[e];
    const int m0  = blockIdx.y * 128;
    if (m0 >= cnt) return;               // uniform early exit
    const int off = offsets[e];
    const int n0  = blockIdx.x * 128;
    const float* W = W1 + (size_t)e * DD * HH;

    __shared__ unsigned Abuf[128 * 16];  // [row][kpair] bf16x2 (low = even K)
    __shared__ unsigned Bbuf[16 * 128];  // [kpair][n]   bf16x2 (low = even K)
    __shared__ int toks[128];

    const int tid = threadIdx.x;
    if (tid < 128) {
        int r = m0 + tid; if (r >= cnt) r = cnt - 1;   // clamp: rows >= cnt never stored
        toks[tid] = list[off + r];
    }
    __syncthreads();

    const int wid = tid >> 5, lane = tid & 31;
    const int hi = lane >> 4, l16 = lane & 15;
    const int wm = (wid & 1) * 64;
    const int wn = (wid >> 1) * 32;

    // --- hoisted, loop-invariant staging bases ---
    const int arow = tid >> 1, ahalf = tid & 1;
    const float* xrow = x + (size_t)(toks[arow] >> 1) * DD + ahalf * 16;
    unsigned* adst = &Abuf[arow * 16 + ahalf * 8];

    const int kp = tid >> 4, seg = tid & 15;
    const float* wb0 = W + (size_t)(2 * kp) * HH + n0 + seg * 8;  // + k0*HH per iter
    unsigned* bdst = &Bbuf[kp * 128 + seg * 8];

    CF acc[4][2];
#pragma unroll
    for (int a = 0; a < 4; ++a)
#pragma unroll
        for (int b = 0; b < 2; ++b)
#pragma unroll
            for (int r = 0; r < 8; ++r) acc[a][b].f[r] = 0.f;

    for (int k0 = 0; k0 < DD; k0 += 32) {
        // ---- stage A (gathered token rows, fp32 -> bf16, b128 LDS stores) ----
        {
            const float* src = xrow + k0;
            float4 f0 = ((const float4*)src)[0];
            float4 f1 = ((const float4*)src)[1];
            float4 f2 = ((const float4*)src)[2];
            float4 f3 = ((const float4*)src)[3];
            if (k0 + 32 < DD) __builtin_prefetch(src + 32, 0, 0);
            adst[0] = packbf(f0.x, f0.y); adst[1] = packbf(f0.z, f0.w);
            adst[2] = packbf(f1.x, f1.y); adst[3] = packbf(f1.z, f1.w);
            adst[4] = packbf(f2.x, f2.y); adst[5] = packbf(f2.z, f2.w);
            adst[6] = packbf(f3.x, f3.y); adst[7] = packbf(f3.z, f3.w);
        }
        // ---- stage B (W1 K-slab, fp32 -> bf16 K-pairs) ----
        {
            const float* r0 = wb0 + (size_t)k0 * HH;
            const float* r1 = r0 + HH;
            if (k0 + 32 < DD) {
                __builtin_prefetch(r0 + 32 * (size_t)HH, 0, 0);
                __builtin_prefetch(r1 + 32 * (size_t)HH, 0, 0);
            }
            stage_B_from(r0, r1, bdst);
        }
        __syncthreads();

        AF a[4], b[2];
#pragma unroll
        for (int mf = 0; mf < 4; ++mf) {
            int row = wm + mf * 16 + l16;
#pragma unroll
            for (int r = 0; r < 8; ++r)   // contiguous K layout (k = 16*hi + s)
                a[mf].u[r] = Abuf[row * 16 + hi * 8 + r];
        }
#pragma unroll
        for (int nf = 0; nf < 2; ++nf) {
            int nc = wn + nf * 16 + l16;
#pragma unroll
            for (int v = 0; v < 8; ++v)   // contiguous K layout (k = 16*hi + s)
                b[nf].u[v] = Bbuf[(v + hi * 8) * 128 + nc];
        }
#pragma unroll
        for (int mf = 0; mf < 4; ++mf)
#pragma unroll
            for (int nf = 0; nf < 2; ++nf)
                acc[mf][nf].v = __builtin_amdgcn_wmma_f32_16x16x32_bf16(
                    false, a[mf].v, false, b[nf].v,
                    (short)0, acc[mf][nf].v, false, false);
        __syncthreads();
    }

    // ---- epilogue: +b1, exact GELU, bf16 store ----
#pragma unroll
    for (int mf = 0; mf < 4; ++mf) {
#pragma unroll
        for (int nf = 0; nf < 2; ++nf) {
            int gn = n0 + wn + nf * 16 + l16;
            float bias = b1[(size_t)e * HH + gn];
#pragma unroll
            for (int r = 0; r < 8; ++r) {
                int gm = m0 + wm + mf * 16 + r + hi * 8;
                if (gm < cnt) {
                    float v = acc[mf][nf].f[r] + bias;
                    float g = 0.5f * v * (1.0f + erff(v * 0.70710678118654752f));
                    hbuf[(size_t)(off + gm) * HH + gn] = f2bfu(g);
                }
            }
        }
    }
}

// ---------------------------------------------------------------------------
// 5) GEMM2: y = (h @ W2[e] + b2[e]) * w  -> yslot[token*2+slot][D]
//    A rows are already bf16 and contiguous -> async copy to LDS (ASYNCcnt)
// ---------------------------------------------------------------------------
__global__ __launch_bounds__(256) void moe_gemm2(
    const unsigned short* __restrict__ hbuf, const float* __restrict__ W2,
    const float* __restrict__ b2,
    const int* __restrict__ counts, const int* __restrict__ offsets,
    const int* __restrict__ list, const float* __restrict__ wlist,
    float* __restrict__ yslot)
{
    const int e   = blockIdx.z;
    const int cnt = counts[e];
    const int m0  = blockIdx.y * 128;
    if (m0 >= cnt) return;
    const int off = offsets[e];
    const int n0  = blockIdx.x * 128;
    const float* W = W2 + (size_t)e * HH * DD;

    __shared__ unsigned Abuf[128 * 16];
    __shared__ unsigned Bbuf[16 * 128];

    const int tid = threadIdx.x;
    const int wid = tid >> 5, lane = tid & 31;
    const int hi = lane >> 4, l16 = lane & 15;
    const int wm = (wid & 1) * 64;
    const int wn = (wid >> 1) * 32;

    // --- hoisted async-copy bases (two b128 transfers per thread per iter) ---
    // LDS byte offset of Abuf: flat shared address low 32 bits == LDS offset.
    const unsigned ldsA = (unsigned)(size_t)(void*)&Abuf[0];
    unsigned long long ga[2];
    unsigned ldoff[2];
#pragma unroll
    for (int it = 0; it < 2; ++it) {
        int idx = tid + it * 256;
        int row = idx >> 2, ch = idx & 3;              // 4 x 16B per 64B row slab
        int gm = m0 + row; if (gm >= cnt) gm = cnt - 1; // clamp: never stored
        ga[it] = (unsigned long long)(size_t)(hbuf + (size_t)(off + gm) * HH + ch * 8);
        ldoff[it] = ldsA + (unsigned)((row * 16 + ch * 4) * 4);
    }
    const int kp = tid >> 4, seg = tid & 15;
    const float* wb0 = W + (size_t)(2 * kp) * DD + n0 + seg * 8;
    unsigned* bdst = &Bbuf[kp * 128 + seg * 8];

    CF acc[4][2];
#pragma unroll
    for (int a = 0; a < 4; ++a)
#pragma unroll
        for (int b = 0; b < 2; ++b)
#pragma unroll
            for (int r = 0; r < 8; ++r) acc[a][b].f[r] = 0.f;

    for (int k0 = 0; k0 < HH; k0 += 32) {
        // ---- stage A: async DMA of contiguous bf16 rows into LDS ----
#pragma unroll
        for (int it = 0; it < 2; ++it) {
            unsigned long long gaddr = ga[it] + (unsigned long long)(k0 * 2);
            asm volatile("global_load_async_to_lds_b128 %0, %1, off"
                         :: "v"(ldoff[it]), "v"(gaddr));
        }
        // ---- stage B: W2 fp32 -> bf16 K-pairs (overlaps with async A) ----
        {
            const float* r0 = wb0 + (size_t)k0 * DD;
            const float* r1 = r0 + DD;
            if (k0 + 32 < HH) {
                __builtin_prefetch(r0 + 32 * (size_t)DD, 0, 0);
                __builtin_prefetch(r1 + 32 * (size_t)DD, 0, 0);
            }
            stage_B_from(r0, r1, bdst);
        }
        asm volatile("s_wait_asynccnt 0x0" ::: "memory");
        __syncthreads();

        AF a[4], b[2];
#pragma unroll
        for (int mf = 0; mf < 4; ++mf) {
            int row = wm + mf * 16 + l16;
#pragma unroll
            for (int r = 0; r < 8; ++r)
                a[mf].u[r] = Abuf[row * 16 + hi * 8 + r];
        }
#pragma unroll
        for (int nf = 0; nf < 2; ++nf) {
            int nc = wn + nf * 16 + l16;
#pragma unroll
            for (int v = 0; v < 8; ++v)
                b[nf].u[v] = Bbuf[(v + hi * 8) * 128 + nc];
        }
#pragma unroll
        for (int mf = 0; mf < 4; ++mf)
#pragma unroll
            for (int nf = 0; nf < 2; ++nf)
                acc[mf][nf].v = __builtin_amdgcn_wmma_f32_16x16x32_bf16(
                    false, a[mf].v, false, b[nf].v,
                    (short)0, acc[mf][nf].v, false, false);
        __syncthreads();
    }

    // ---- epilogue: +b2, * combine weight, scatter to (token,slot) rows ----
#pragma unroll
    for (int mf = 0; mf < 4; ++mf) {
#pragma unroll
        for (int r = 0; r < 8; ++r) {
            int gm = m0 + wm + mf * 16 + r + hi * 8;
            if (gm < cnt) {
                int   packed = list[off + gm];
                float w      = wlist[off + gm];
#pragma unroll
                for (int nf = 0; nf < 2; ++nf) {
                    int gn = n0 + wn + nf * 16 + l16;
                    float v = acc[mf][nf].f[r] + b2[(size_t)e * DD + gn];
                    yslot[(size_t)packed * DD + gn] = w * v;
                }
            }
        }
    }
}

// ---------------------------------------------------------------------------
// 6) combine: out[t] = yslot[2t] + yslot[2t+1]   (float4 traffic)
// ---------------------------------------------------------------------------
__global__ __launch_bounds__(256) void moe_combine(
    const float* __restrict__ yslot, float* __restrict__ out)
{
    size_t gi = (size_t)blockIdx.x * 256 + threadIdx.x;  // float4 index
    size_t t = gi >> 8;        // 256 float4 per token row
    size_t c = gi & 255;
    const float4* ys = (const float4*)yslot;
    float4 a = ys[(t * 2) * 256 + c];
    float4 b = ys[(t * 2 + 1) * 256 + c];
    float4 o;
    o.x = a.x + b.x; o.y = a.y + b.y; o.z = a.z + b.z; o.w = a.w + b.w;
    ((float4*)out)[t * 256 + c] = o;
}

// ---------------------------------------------------------------------------
extern "C" void kernel_launch(void* const* d_in, const int* in_sizes, int n_in,
                              void* d_out, int out_size, void* d_ws, size_t ws_size,
                              hipStream_t stream)
{
    (void)in_sizes; (void)n_in; (void)out_size; (void)ws_size;
    const float* x  = (const float*)d_in[0];
    const float* Wg = (const float*)d_in[1];
    const float* bg = (const float*)d_in[2];
    const float* W1 = (const float*)d_in[3];
    const float* b1 = (const float*)d_in[4];
    const float* W2 = (const float*)d_in[5];
    const float* b2 = (const float*)d_in[6];
    float* out = (float*)d_out;

    char* ws = (char*)d_ws;
    int*   counts  = (int*)(ws + 0);
    int*   cursors = (int*)(ws + 64);
    int*   offsets = (int*)(ws + 128);
    int*   topke   = (int*)(ws + 256);
    float* w0a     = (float*)(ws + 256 + 4 * NTOK);
    float* w1a     = (float*)(ws + 256 + 8 * NTOK);
    float* gp      = (float*)(ws + 256 + 12 * NTOK);
    int*   list    = (int*)(ws + (256 << 10));
    float* wlist   = (float*)(ws + (256 << 10) + 4 * NSLOT);
    unsigned short* hbuf = (unsigned short*)(ws + (1 << 20));
    float* yslot   = (float*)(ws + (1 << 20) + (size_t)2 * NSLOT * HH);

    moe_init<<<1, 64, 0, stream>>>(counts, cursors);
    moe_gate<<<NTOK / 8, 256, 0, stream>>>(x, Wg, bg, counts, topke, w0a, w1a, gp);
    moe_scan<<<1, 1, 0, stream>>>(counts, offsets, gp, out + (size_t)NTOK * DD);
    moe_build<<<NTOK / 256, 256, 0, stream>>>(topke, w0a, w1a, offsets, cursors, list, wlist);
    moe_gemm1<<<dim3(HH / 128, 64, EE), 256, 0, stream>>>(x, W1, b1, counts, offsets, list, hbuf);
    moe_gemm2<<<dim3(DD / 128, 64, EE), 256, 0, stream>>>(hbuf, W2, b2, counts, offsets, list,
                                                          wlist, yslot);
    moe_combine<<<NTOK, 256, 0, stream>>>(yslot, out);
}
